// QuantizedE8SLinear_45990509806223
// MI455X (gfx1250) — matrix operand order repaired and checked by reading, not compiled, not run on hardware
//
#include <hip/hip_runtime.h>

typedef __attribute__((ext_vector_type(16))) _Float16 v16h;
typedef __attribute__((ext_vector_type(8)))  _Float16 v8h;
typedef __attribute__((ext_vector_type(8)))  float    v8f;
typedef int b128_t __attribute__((vector_size(16)));   // matches builtin param type

#define M_TOTAL 8192      // B*S rows of x
#define N_TOTAL 4096      // m (rows of W)
#define K_TOTAL 4096      // n * CODESZ
#define CODES_PER_ROW 512
#define BK 64             // K-step through LDS
#define LSTR 72           // LDS row stride in halfs (64 + 8 pad; 144B, 16B-aligned, conflict-free)

#define AS1 __attribute__((address_space(1)))
#define AS3 __attribute__((address_space(3)))

#if __has_builtin(__builtin_amdgcn_global_load_async_to_lds_b128)
#define HAVE_ASYNC_LDS 1
#endif

// 16-byte global -> LDS copy; async DMA path on CDNA5 when available.
__device__ __forceinline__ void copy16_g2l(const _Float16* g, _Float16* l) {
#if defined(HAVE_ASYNC_LDS)
  __builtin_amdgcn_global_load_async_to_lds_b128(
      (AS1 b128_t*)(g), (AS3 b128_t*)(l), /*offset=*/0, /*cpol=*/0);
#else
  *(v8h*)l = *(const v8h*)g;
#endif
}

__device__ __forceinline__ void wait_g2l() {
#if defined(HAVE_ASYNC_LDS)
#if __has_builtin(__builtin_amdgcn_s_wait_asynccnt)
  __builtin_amdgcn_s_wait_asynccnt(0);
#else
  asm volatile("s_wait_asynccnt 0" ::: "memory");
#endif
#endif
}

// ---------------- Kernel 1: decode E8 codebook -> Wh (fp16, 4096x4096) ----
__global__ void decode_w(const int* __restrict__ qidxs,
                         const _Float16* __restrict__ grid,
                         _Float16* __restrict__ wh) {
  int t = blockIdx.x * blockDim.x + threadIdx.x;   // j*512 + c
  int code = qidxs[t];
  const v8h* g = (const v8h*)grid;                 // 8 halfs = 16B per code
  ((v8h*)wh)[t] = g[code];                         // halfs offset = j*4096 + c*8
}

// ---------------- Kernel 2: x*SU -> FWHT(4096) -> /65536 -> fp16 ----------
__global__ void fwht_in(const float* __restrict__ inp,
                        const float* __restrict__ su,
                        _Float16* __restrict__ xh) {
  __shared__ float l[4096];
  const int row = blockIdx.x;
  const int tid = threadIdx.x;
  const float* x = inp + (size_t)row * K_TOTAL;
#pragma unroll
  for (int p = 0; p < 16; p++) { int k = tid + p * 256; l[k] = x[k] * su[k]; }
  for (int h = 1; h < 4096; h <<= 1) {
    __syncthreads();
    for (int t = tid; t < 2048; t += 256) {
      int i = ((t & ~(h - 1)) << 1) | (t & (h - 1));
      int j = i + h;
      float a = l[i], b = l[j];
      l[i] = a + b; l[j] = a - b;
    }
  }
  __syncthreads();
  _Float16* y = xh + (size_t)row * K_TOTAL;
  const float s = 1.0f / 65536.0f;                 // (1/sqrt(4096)) * (1/1024)
#pragma unroll
  for (int p = 0; p < 16; p++) { int k = tid + p * 256; y[k] = (_Float16)(l[k] * s); }
}

// ---------------- Kernel 3: z = Xh @ Wh^T via v_wmma_f32_16x16x32_f16 -----
// 128x128 tile per 256-thread block; 8 waves in 2x4 layout, 4x2 16x16 tiles
// per wave; K stepped by 64 through double-buffered LDS (async copy on CDNA5).
__launch_bounds__(256)
__global__ void gemm_wmma(const _Float16* __restrict__ xh,
                          const _Float16* __restrict__ wh,
                          float* __restrict__ zout) {
  __shared__ _Float16 lA[2][128 * LSTR];
  __shared__ _Float16 lB[2][128 * LSTR];
  const int tid    = threadIdx.x;
  const int wave   = tid >> 5;
  const int lane   = tid & 31;
  const int laneLo = lane & 15;
  const int hi     = lane >> 4;          // lane 16..31 -> high half of wave
  const int wm     = wave & 1;           // 2 wave rows  (64 M each)
  const int wn     = wave >> 1;          // 4 wave cols  (32 N each)
  const int rowBase = blockIdx.x * 128;  // M tile base (x rows)
  const int colBase = blockIdx.y * 128;  // N tile base (W rows)

  // per-thread staging coords: 4 chunks of 8 halfs per matrix per K-step
  int srow[4], scol[4];
#pragma unroll
  for (int p = 0; p < 4; p++) {
    int i = tid + p * 256;
    srow[p] = i >> 3;                    // tile row 0..127
    scol[p] = (i & 7) << 3;              // half offset 0..56
  }

  v8f acc[4][2] = {};                    // 4 M-subtiles x 2 N-subtiles, f32 16x16 each

  // ---- prologue: stage K-tile 0 into buffer 0
#pragma unroll
  for (int p = 0; p < 4; p++) {
    copy16_g2l(&xh[(size_t)(rowBase + srow[p]) * K_TOTAL + scol[p]],
               &lA[0][srow[p] * LSTR + scol[p]]);
    copy16_g2l(&wh[(size_t)(colBase + srow[p]) * K_TOTAL + scol[p]],
               &lB[0][srow[p] * LSTR + scol[p]]);
  }

  int cur = 0;
  for (int kb = 0; kb < K_TOTAL; kb += BK) {
    wait_g2l();
    __syncthreads();                     // buffer `cur` ready for everyone

    // stage next K-tile into the other buffer while we compute
    if (kb + BK < K_TOTAL) {
      const int nb = kb + BK;
      const int nxt = cur ^ 1;
#pragma unroll
      for (int p = 0; p < 4; p++) {
        copy16_g2l(&xh[(size_t)(rowBase + srow[p]) * K_TOTAL + nb + scol[p]],
                   &lA[nxt][srow[p] * LSTR + scol[p]]);
        copy16_g2l(&wh[(size_t)(colBase + srow[p]) * K_TOTAL + nb + scol[p]],
                   &lB[nxt][srow[p] * LSTR + scol[p]]);
      }
    }

    const _Float16* A = lA[cur];
    const _Float16* B = lB[cur];
#pragma unroll
    for (int kk = 0; kk < 2; kk++) {     // two K=32 slices per LDS tile
      const int kof = kk * 32;
      // preload ALL fragments for this slice, then issue WMMAs back-to-back
      v16h bf[2];
#pragma unroll
      for (int nt = 0; nt < 2; nt++) {   // B: 32x16, lane: n=laneLo, k=hi*16+0..15
        const int br = wn * 32 + nt * 16 + laneLo;
        const v8h b0 = *(const v8h*)&B[br * LSTR + kof + hi * 16];
        const v8h b1 = *(const v8h*)&B[br * LSTR + kof + hi * 16 + 8];
#pragma unroll
        for (int e = 0; e < 8; e++) { bf[nt][e] = b0[e]; bf[nt][e + 8] = b1[e]; }
      }
      v16h af[4];
#pragma unroll
      for (int mt = 0; mt < 4; mt++) {   // A: 16x32, lane: m=laneLo, k=hi*16+0..15
        const int ar = wm * 64 + mt * 16 + laneLo;
        const v8h a0 = *(const v8h*)&A[ar * LSTR + kof + hi * 16];
        const v8h a1 = *(const v8h*)&A[ar * LSTR + kof + hi * 16 + 8];
#pragma unroll
        for (int e = 0; e < 8; e++) { af[mt][e] = a0[e]; af[mt][e + 8] = a1[e]; }
      }
#pragma unroll
      for (int mt = 0; mt < 4; mt++)
#pragma unroll
        for (int nt = 0; nt < 2; nt++)
          acc[mt][nt] = __builtin_amdgcn_wmma_f32_16x16x32_f16(
              /*neg_a=*/false, af[mt], /*neg_b=*/false, bf[nt],
              /*c_mod=*/(short)0, acc[mt][nt], /*reuse_a=*/false, /*reuse_b=*/false);
    }
    cur ^= 1;
  }

  // ---- store f32 z tile: C/D layout -> m = base + r + hi*8, n = base + laneLo
#pragma unroll
  for (int mt = 0; mt < 4; mt++) {
    const int mrow = rowBase + wm * 64 + mt * 16 + hi * 8;
#pragma unroll
    for (int nt = 0; nt < 2; nt++) {
      const int ncol = colBase + wn * 32 + nt * 16 + laneLo;
#pragma unroll
      for (int r = 0; r < 8; r++) {
        zout[(size_t)(mrow + r) * N_TOTAL + ncol] = acc[mt][nt][r];
      }
    }
  }
}

// ---------------- Kernel 4: y = FWHT(z*Wscale*1024)/64 * SV  (in-place) ---
__global__ void fwht_out(float* __restrict__ out,
                         const float* __restrict__ sv,
                         const float* __restrict__ wscale) {
  __shared__ float l[4096];
  const int row = blockIdx.x;
  const int tid = threadIdx.x;
  const float s = wscale[0] * 1024.0f;
  float* y = out + (size_t)row * N_TOTAL;
#pragma unroll
  for (int p = 0; p < 16; p++) { int k = tid + p * 256; l[k] = y[k] * s; }
  for (int h = 1; h < 4096; h <<= 1) {
    __syncthreads();
    for (int t = tid; t < 2048; t += 256) {
      int i = ((t & ~(h - 1)) << 1) | (t & (h - 1));
      int j = i + h;
      float a = l[i], b = l[j];
      l[i] = a + b; l[j] = a - b;
    }
  }
  __syncthreads();
#pragma unroll
  for (int p = 0; p < 16; p++) {
    int k = tid + p * 256;
    y[k] = l[k] * 0.015625f * sv[k];     // 1/sqrt(4096) = 1/64
  }
}

extern "C" void kernel_launch(void* const* d_in, const int* in_sizes, int n_in,
                              void* d_out, int out_size, void* d_ws, size_t ws_size,
                              hipStream_t stream) {
  const float*    inp    = (const float*)d_in[0];     // (4,2048,4096) f32
  const float*    su     = (const float*)d_in[1];     // (4096,) f32
  const float*    sv     = (const float*)d_in[2];     // (4096,) f32
  const float*    wscale = (const float*)d_in[3];     // scalar f32
  const _Float16* grid   = (const _Float16*)d_in[4];  // (ncodes, 8) f16
  const int*      qidxs  = (const int*)d_in[5];       // (4096, 512) i32
  float*          out    = (float*)d_out;             // (4,2048,4096) f32

  _Float16* wh = (_Float16*)d_ws;                                          // 32 MB
  _Float16* xh = (_Float16*)((char*)d_ws + (size_t)N_TOTAL * K_TOTAL * 2); // 64 MB

  decode_w<<<(N_TOTAL * CODES_PER_ROW) / 256, 256, 0, stream>>>(qidxs, grid, wh);
  fwht_in<<<M_TOTAL, 256, 0, stream>>>(inp, su, xh);
  gemm_wmma<<<dim3(M_TOTAL / 128, N_TOTAL / 128), 256, 0, stream>>>(xh, wh, out);
  fwht_out<<<M_TOTAL, 256, 0, stream>>>(out, sv, wscale);
}
